// Transformer_42425686950259
// MI455X (gfx1250) — matrix operand order repaired and checked by reading, not compiled, not run on hardware
//
#include <hip/hip_runtime.h>
#include <hip/hip_bf16.h>
#include <math.h>

// ---------------- model constants ----------------
#define NN      16384      // N nodes
#define IN_DIM  16
#define DH      64         // dim per head
#define HH      8          // heads
#define HD      512        // H*D
#define LL      4          // layers
#define BSZ     64         // attention block size
#define NB      256        // NN / BSZ
#define NHASH   2
#define MM      16         // rpe feat dim
#define EPAD    96         // 64 + 16 + 2 -> padded to mult of 32
#define CATW    320        // (L+1)*D
#define HALFW   32
#define MLPH    256
#define OUTW    32

#define ACT_NONE 0
#define ACT_RELU 1
#define ACT_TANH 2

typedef __attribute__((ext_vector_type(16))) _Float16 v16h;
typedef __attribute__((ext_vector_type(8)))  _Float16 v8h;
typedef __attribute__((ext_vector_type(8)))  float    v8f;

// WMMA 16x16x32 f16 layouts:
// A (16x32, row = lane&15): lane<16 -> K {0..15}; lane>=16 -> K {16..31} (contiguous order used in this file)
// B (32x16, col = lane&15): lane<16 -> K {0..15};       lane>=16 -> K {16..31} (same contiguous order as A, chosen by this code)
// C/D: elem i -> row = i + (lane&16?8:0), col = lane&15

__device__ __forceinline__ v16h cat8(v8h lo, v8h hi) {
  v16h r;
#pragma unroll
  for (int e = 0; e < 8; ++e) { r[e] = lo[e]; r[e + 8] = hi[e]; }
  return r;
}

// A fragment from row-major f32 (K % 32 == 0): 4 unconditional float4 loads
__device__ __forceinline__ v16h load_a_f32(const float* A, int lda, int row0,
                                           int k0, int lane) {
  int r  = row0 + (lane & 15);
  int kb = k0 + ((lane & 16) ? 16 : 0);
  const float4* p0 = (const float4*)(A + (size_t)r * lda + kb);
  const float4* p1 = (const float4*)(A + (size_t)r * lda + kb + 8);
  float4 f0 = p0[0], f1 = p0[1], f2 = p1[0], f3 = p1[1];
  v16h a;
  a[0]  = (_Float16)f0.x; a[1]  = (_Float16)f0.y; a[2]  = (_Float16)f0.z; a[3]  = (_Float16)f0.w;
  a[4]  = (_Float16)f1.x; a[5]  = (_Float16)f1.y; a[6]  = (_Float16)f1.z; a[7]  = (_Float16)f1.w;
  a[8]  = (_Float16)f2.x; a[9]  = (_Float16)f2.y; a[10] = (_Float16)f2.z; a[11] = (_Float16)f2.w;
  a[12] = (_Float16)f3.x; a[13] = (_Float16)f3.y; a[14] = (_Float16)f3.z; a[15] = (_Float16)f3.w;
  return a;
}

// CDNA5 async DMA: 16 bytes/lane, global -> LDS, tracked by ASYNCcnt
__device__ __forceinline__ void async_ld_b128(unsigned lds_addr, unsigned long long gaddr) {
  asm volatile("global_load_async_to_lds_b128 %0, %1, off"
               :: "v"(lds_addr), "v"(gaddr) : "memory");
}
__device__ __forceinline__ void wait_async0() {
  asm volatile("s_wait_asynccnt 0" ::: "memory");
}

// ---------------- weight pack: B[K,Nc] f32 -> fragment-major f16 ----------------
// layout: [tn][kblk][lane][16 halves]  => one aligned v16h load per B fragment
__global__ void pack_b_kernel(const float* __restrict__ B, _Float16* __restrict__ Bp,
                              int K, int Nc) {
  int idx = blockIdx.x * blockDim.x + threadIdx.x;
  int kblks = K >> 5;
  int total = (Nc >> 4) * kblks * 32;
  if (idx >= total) return;
  int lane = idx & 31;
  int blk  = idx >> 5;               // tn*kblks + kblk
  int kblk = blk % kblks;
  int tn   = blk / kblks;
  int c    = tn * 16 + (lane & 15);
  int kb   = kblk * 32 + ((lane & 16) ? 16 : 0);
  v16h v;
#pragma unroll
  for (int e = 0; e < 16; ++e)
    v[e] = (_Float16)B[(size_t)(kb + e) * Nc + c];
  *(v16h*)(Bp + ((size_t)blk * 32 + lane) * 16) = v;
}

// ---------------- generic GEMM: C = act(alpha*(A@Bp) + bias [+ residual]) ----------------
// outf16 != 0 -> Cout is _Float16*, else float*
__global__ void __launch_bounds__(128)
gemm_wmma(const float* __restrict__ A, const _Float16* __restrict__ Bp,
          const float* __restrict__ bias, const float* __restrict__ residual,
          void* __restrict__ Cout, int M, int K, int Nc, float alpha, int act,
          int outf16) {
  int lane = threadIdx.x & 31;
  int wave = threadIdx.x >> 5;
  int tilesM = M >> 4, tilesN = Nc >> 4;
  int tile = blockIdx.x * 4 + wave;
  if (tile >= tilesM * tilesN) return;     // wave-uniform: EXEC stays all-ones
  int tm = tile % tilesM, tn = tile / tilesM;
  int kblks = K >> 5;
  const v16h* bfrag = (const v16h*)Bp + ((size_t)tn * kblks) * 32 + lane;
  v8f acc = {};
  for (int kb = 0; kb < kblks; ++kb) {
    v16h a = load_a_f32(A, K, tm * 16, kb * 32, lane);
    v16h b = bfrag[(size_t)kb * 32];
    acc = __builtin_amdgcn_wmma_f32_16x16x32_f16(false, a, false, b,
                                                 (short)0, acc, false, false);
  }
  int col = tn * 16 + (lane & 15);
  float bv = bias ? bias[col] : 0.0f;
#pragma unroll
  for (int i = 0; i < 8; ++i) {
    int row = tm * 16 + i + ((lane & 16) ? 8 : 0);
    float v = alpha * acc[i] + bv;
    if (residual) v += residual[(size_t)row * Nc + col];
    if (act == ACT_RELU)      v = fmaxf(v, 0.0f);
    else if (act == ACT_TANH) v = tanhf(v);
    if (outf16) ((_Float16*)Cout)[(size_t)row * Nc + col] = (_Float16)v;
    else        ((float*)Cout)[(size_t)row * Nc + col] = v;
  }
}

// ---------------- bitonic sort (in LDS, 64-bit keys = key<<32 | idx => stable) ------------
__device__ __forceinline__ void bitonic_sort_lds(unsigned long long* keys, int tid) {
  for (int k = 2; k <= NN; k <<= 1) {
    for (int j = k >> 1; j > 0; j >>= 1) {
      for (int i = tid; i < NN; i += 1024) {
        int ixj = i ^ j;
        if (ixj > i) {
          unsigned long long a = keys[i], b = keys[ixj];
          bool up = ((i & k) == 0);
          bool sw = up ? (a > b) : (a < b);
          if (sw) { keys[i] = b; keys[ixj] = a; }
        }
      }
      __syncthreads();
    }
  }
}

// ranks of coords[:,0] and coords[:,1]  (blockIdx.x = column)
__global__ void __launch_bounds__(1024)
rank_sort_kernel(const float* __restrict__ coords, int* __restrict__ ranks) {
  extern __shared__ unsigned long long keys[];
  int col = blockIdx.x, tid = threadIdx.x;
  for (int i = tid; i < NN; i += 1024) {
    unsigned u = __float_as_uint(coords[(size_t)i * 3 + col]);
    u = (u & 0x80000000u) ? ~u : (u | 0x80000000u);   // order-preserving map
    keys[i] = ((unsigned long long)u << 32) | (unsigned)i;
  }
  __syncthreads();
  bitonic_sort_lds(keys, tid);
  for (int i = tid; i < NN; i += 1024)
    ranks[(size_t)col * NN + (unsigned)(keys[i] & 0xffffffffu)] = i;
}

// bucket keys + stable argsort -> order   (blockIdx.x = nh*8 + h)
// batch==0 so the second bit_shift is identity; max(reg_e) is closed-form
// because ranks are a permutation 0..N-1 (max attained at rank N-1).
__global__ void __launch_bounds__(1024)
order_sort_kernel(const int* __restrict__ ranks, const float* __restrict__ regions,
                  int* __restrict__ order) {
  extern __shared__ unsigned long long keys[];
  int nh = blockIdx.x >> 3, h = blockIdx.x & 7, tid = threadIdx.x;
  float re = regions[(nh * 2 + 0) * HH + h];
  float rp = regions[(nh * 2 + 1) * HH + h];
  float qe = (float)NN / re, qp = (float)NN / rp;
  int mxe = (int)floorf((float)(NN - 1) / qe) + 1;
  int nb  = (int)ceilf(log2f((float)mxe + 1.0f));
  for (int i = tid; i < NN; i += 1024) {
    int reg_e = (int)floorf((float)ranks[i] / qe) + 1;
    int reg_p = (int)floorf((float)ranks[NN + i] / qp) + 1;
    unsigned key = ((unsigned)reg_p << nb) | (unsigned)reg_e;
    keys[i] = ((unsigned long long)key << 32) | (unsigned)i;
  }
  __syncthreads();
  bitonic_sort_lds(keys, tid);
  size_t base = ((size_t)nh * HH + h) * NN;
  for (int i = tid; i < NN; i += 1024)
    order[base + i] = (int)(keys[i] & 0xffffffffu);
}

// ---------------- layernorm (one wave per row), optional tanh ----------------
__global__ void __launch_bounds__(256)
layernorm_kernel(const float* __restrict__ x, const float* __restrict__ g,
                 const float* __restrict__ b, float* __restrict__ y,
                 int n, int W, int act) {
  int wv = (blockIdx.x * blockDim.x + threadIdx.x) >> 5;
  int lane = threadIdx.x & 31;
  if (wv >= n) return;
  const float* row = x + (size_t)wv * W;
  float s = 0.f;
  for (int c = lane; c < W; c += 32) s += row[c];
  for (int o = 16; o; o >>= 1) s += __shfl_xor(s, o, 32);
  float mu = s / (float)W;
  float v = 0.f;
  for (int c = lane; c < W; c += 32) { float d = row[c] - mu; v += d * d; }
  for (int o = 16; o; o >>= 1) v += __shfl_xor(v, o, 32);
  float rs = rsqrtf(v / (float)W + 1e-5f);
  for (int c = lane; c < W; c += 32) {
    float o = (row[c] - mu) * rs * g[c] + b[c];
    if (act == ACT_TANH) o = tanhf(o);
    y[(size_t)wv * W + c] = o;
  }
}

// ---------------- rpe helpers ----------------
__global__ void omega_kernel(const float* __restrict__ w_rpe, float* __restrict__ omega) {
  int idx = blockIdx.x * blockDim.x + threadIdx.x;       // H*M = 128
  if (idx >= HH * MM) return;
  int h = idx >> 4, m = idx & 15;
  float s = 0.f;
  for (int d = 0; d < DH; ++d) s += w_rpe[(size_t)(h * DH + d) * MM + m];
  omega[idx] = s / (float)DH;
}

__global__ void sq_kernel(const float* __restrict__ coords, const float* __restrict__ omega,
                          float* __restrict__ sq) {
  int idx = blockIdx.x * blockDim.x + threadIdx.x;       // N*H
  if (idx >= NN * HH) return;
  int h = idx & 7, n = idx >> 3;
  float c0 = coords[(size_t)n * 3], c1 = coords[(size_t)n * 3 + 1];
  float s = 0.f;
#pragma unroll
  for (int m = 0; m < MM; ++m) {
    float w = (m < 8 ? c0 : c1) * omega[h * MM + m];
    s += w * w;
  }
  sq[idx] = s;
}

// ---------------- fused per-block attention ----------------
// blockIdx.x = (nh*H + h)*NB + b ; 256 threads (8 waves)
// q16/k16 are f16 [N][HD]; their 64-col rows are DMA'd straight into LDS tiles.
__global__ void __launch_bounds__(256)
attn_block_kernel(const _Float16* __restrict__ q16, const _Float16* __restrict__ k16,
                  const float* __restrict__ v, const float* __restrict__ coords,
                  const float* __restrict__ omega, const float* __restrict__ sq,
                  const int* __restrict__ order,
                  float* __restrict__ out_r, float* __restrict__ lse_r) {
  __shared__ alignas(32) _Float16 qs[BSZ * EPAD];   // later reused for P (stride BSZ)
  __shared__ alignas(32) _Float16 ks[BSZ * EPAD];   // row-major [n][e] == B-frag contiguous
  __shared__ alignas(32) _Float16 vsT[DH * BSZ];    // transposed: [d][k] == B-frag contiguous
  __shared__ float    sS[BSZ * BSZ];
  __shared__ int      node_s[BSZ];
  __shared__ float    c0s[BSZ], c1s[BSZ], sqs[BSZ], om[MM];

  int tid  = threadIdx.x;
  int b    = blockIdx.x & (NB - 1);
  int h    = (blockIdx.x >> 8) & 7;
  int nh   = blockIdx.x >> 11;
  size_t obase = ((size_t)nh * HH + h) * NN;

  if (tid < BSZ) {
    int n = order[obase + (size_t)b * BSZ + tid];
    node_s[tid] = n;
    c0s[tid] = coords[(size_t)n * 3];
    c1s[tid] = coords[(size_t)n * 3 + 1];
    sqs[tid] = sq[(size_t)n * HH + h];
  }
  if (tid < MM) om[tid] = omega[h * MM + tid];
  __syncthreads();

  // ---- kick off async DMA of q/k rows (cols 0..63, f16) into qs/ks ----
  // 64 rows x 8 x b128 x {q,k} = 1024 lane-transfers; 4 per thread
#pragma unroll
  for (int it = 0; it < 4; ++it) {
    int idx  = it * 256 + tid;
    int mat  = idx >> 9;               // 0 = q, 1 = k
    int rem  = idx & 511;
    int r    = rem >> 3;
    int part = rem & 7;                // 8 halves = 16 bytes
    const _Float16* src = (mat ? k16 : q16)
        + (size_t)node_s[r] * HD + h * DH + part * 8;
    _Float16* dst = (mat ? ks : qs) + r * EPAD + part * 8;
    async_ld_b128((unsigned)(unsigned long long)dst, (unsigned long long)src);
  }

  // ---- overlap: build the RPE tail (cols 64..95) and V^T while DMA flies ----
  {
    int r = tid >> 2, seg = tid & 3;
    const float s2 = 1.41421356237f;
    for (int c = DH + seg * 8; c < DH + seg * 8 + 8; ++c) {
      float qv, kv;
      if (c < DH + MM) {
        int m = c - DH;
        float w = (m < 8 ? c0s[r] : c1s[r]) * om[m] * s2;
        qv = w; kv = w;
      } else if (c == DH + MM)     { qv = -sqs[r]; kv = 1.0f; }
      else if (c == DH + MM + 1)   { qv = 1.0f;    kv = -sqs[r]; }
      else                         { qv = 0.0f;    kv = 0.0f; }
      qs[r * EPAD + c] = (_Float16)qv;
      ks[r * EPAD + c] = (_Float16)kv;
    }
    int n = node_s[r];
    for (int c = seg * 16; c < seg * 16 + 16; ++c)
      vsT[c * BSZ + r] = (_Float16)v[(size_t)n * HD + h * DH + c];
  }
  wait_async0();      // this wave's DMA into LDS retired
  __syncthreads();    // all waves' tiles visible

  int lane = tid & 31, wave = tid >> 5;

  // S = Q_aug @ K_aug^T : 16 tiles of 16x16, K loop over EPAD
  for (int t = wave; t < 16; t += 8) {
    int ti = t & 3, tj = t >> 2;
    v8f acc = {};
#pragma unroll
    for (int k0 = 0; k0 < EPAD; k0 += 32) {
      int rr = ti * 16 + (lane & 15);
      int ka = k0 + ((lane & 16) ? 16 : 0);
      v16h a = cat8(*(const v8h*)&qs[rr * EPAD + ka],
                    *(const v8h*)&qs[rr * EPAD + ka + 8]);
      int cc = tj * 16 + (lane & 15);
      int kbb = k0 + ((lane & 16) ? 16 : 0);
      v16h bf = *(const v16h*)&ks[cc * EPAD + kbb];   // B[k][n] = ks[n][k], contiguous in k
      acc = __builtin_amdgcn_wmma_f32_16x16x32_f16(false, a, false, bf,
                                                   (short)0, acc, false, false);
    }
    int col = tj * 16 + (lane & 15);
#pragma unroll
    for (int i = 0; i < 8; ++i) {
      int row = ti * 16 + i + ((lane & 16) ? 8 : 0);
      sS[row * BSZ + col] = acc[i];
    }
  }
  __syncthreads();

  // rowwise logsumexp-softmax; write P (f16) into qs buffer; scatter lse
  {
    int r = tid >> 2, seg = tid & 3;
    float mx = -1e30f;
    for (int c = seg * 16; c < seg * 16 + 16; ++c) mx = fmaxf(mx, sS[r * BSZ + c]);
    mx = fmaxf(mx, __shfl_xor(mx, 1, 32));
    mx = fmaxf(mx, __shfl_xor(mx, 2, 32));
    float sum = 0.f;
    for (int c = seg * 16; c < seg * 16 + 16; ++c) sum += __expf(sS[r * BSZ + c] - mx);
    sum += __shfl_xor(sum, 1, 32);
    sum += __shfl_xor(sum, 2, 32);
    float lse = mx + __logf(sum);
    for (int c = seg * 16; c < seg * 16 + 16; ++c)
      qs[r * BSZ + c] = (_Float16)__expf(sS[r * BSZ + c] - lse);
    if (seg == 0) lse_r[obase + node_s[r]] = lse;
  }
  __syncthreads();

  // O = P @ V : 16 tiles, K = 64 ; scatter rows to node positions (== inv gather)
  for (int t = wave; t < 16; t += 8) {
    int ti = t & 3, tj = t >> 2;
    v8f acc = {};
#pragma unroll
    for (int k0 = 0; k0 < BSZ; k0 += 32) {
      int rr = ti * 16 + (lane & 15);
      int ka = k0 + ((lane & 16) ? 16 : 0);
      v16h a = cat8(*(const v8h*)&qs[rr * BSZ + ka],
                    *(const v8h*)&qs[rr * BSZ + ka + 8]);
      int cc = tj * 16 + (lane & 15);
      int kbb = k0 + ((lane & 16) ? 16 : 0);
      v16h bf = *(const v16h*)&vsT[cc * BSZ + kbb];   // B[k][n] = vsT[n][k], contiguous in k
      acc = __builtin_amdgcn_wmma_f32_16x16x32_f16(false, a, false, bf,
                                                   (short)0, acc, false, false);
    }
    int col = tj * 16 + (lane & 15);
#pragma unroll
    for (int i = 0; i < 8; ++i) {
      int row = ti * 16 + i + ((lane & 16) ? 8 : 0);
      int n = node_s[row];
      out_r[(obase + n) * DH + col] = acc[i];
    }
  }
}

// ---------------- combine hash rounds by lse softmax ----------------
__global__ void combine_rounds(const float* __restrict__ out_r,
                               const float* __restrict__ lse_r,
                               float* __restrict__ agg) {
  size_t idx = (size_t)blockIdx.x * blockDim.x + threadIdx.x;
  if (idx >= (size_t)NN * HD) return;
  int d = idx & 63;
  int h = (int)((idx >> 6) & 7);
  size_t n = idx >> 9;
  size_t p0 = ((size_t)0 * HH + h) * NN + n;
  size_t p1 = ((size_t)1 * HH + h) * NN + n;
  float l0 = lse_r[p0], l1 = lse_r[p1];
  float m  = fmaxf(l0, l1);
  float e0 = __expf(l0 - m), e1 = __expf(l1 - m);
  float w  = 1.0f / (e0 + e1);
  agg[n * HD + (size_t)h * DH + d] =
      (e0 * out_r[p0 * DH + d] + e1 * out_r[p1 * DH + d]) * w;
}

// ---------------- misc small kernels ----------------
__global__ void pad_x_kernel(const float* __restrict__ x, float* __restrict__ xp) {
  int idx = blockIdx.x * blockDim.x + threadIdx.x;       // NN*32
  if (idx >= NN * 32) return;
  int c = idx & 31, n = idx >> 5;
  xp[idx] = (c < IN_DIM) ? x[n * IN_DIM + c] : 0.0f;
}

__global__ void copy_cols(const float* __restrict__ src, float* __restrict__ dst, int colofs) {
  size_t idx = (size_t)blockIdx.x * blockDim.x + threadIdx.x;
  if (idx >= (size_t)NN * DH) return;
  size_t n = idx >> 6; int c = (int)(idx & 63);
  dst[n * CATW + colofs + c] = src[idx];
}

__global__ void zero_kernel(float* p, int n) {
  int i = blockIdx.x * blockDim.x + threadIdx.x;
  if (i < n) p[i] = 0.f;
}

__global__ void __launch_bounds__(256)
colsum_kernel(const float* __restrict__ t, float* __restrict__ accum) {
  __shared__ float part[HALFW];
  if (threadIdx.x < HALFW) part[threadIdx.x] = 0.f;
  __syncthreads();
  int rows = NN / gridDim.x;
  int r0 = blockIdx.x * rows;
  int c = threadIdx.x & 31;
  float s = 0.f;
  for (int r = r0 + (threadIdx.x >> 5); r < r0 + rows; r += 8)
    s += t[(size_t)r * HALFW + c];
  atomicAdd(&part[c], s);
  __syncthreads();
  if (threadIdx.x < HALFW) atomicAdd(&accum[threadIdx.x], part[threadIdx.x]);
}

__global__ void final_kernel(const float* __restrict__ accum, const float* __restrict__ pw,
                             const float* __restrict__ pb, float* __restrict__ out) {
  int o = threadIdx.x;
  if (o >= OUTW) return;
  float s = 0.f;
  for (int i = 0; i < HALFW; ++i)
    s += (accum[i] / (float)NN) * pw[i * OUTW + o];
  out[o] = s + pb[o];
}

// ---------------- host orchestration ----------------
extern "C" void kernel_launch(void* const* d_in, const int* in_sizes, int n_in,
                              void* d_out, int out_size, void* d_ws, size_t ws_size,
                              hipStream_t stream) {
  (void)in_sizes; (void)n_in; (void)out_size; (void)ws_size;
  const float* x       = (const float*)d_in[0];
  const float* coords  = (const float*)d_in[1];
  /* batch (d_in[2]) is all zeros -> unused */
  const float* regions = (const float*)d_in[3];
  const float* fe_w1   = (const float*)d_in[4];
  const float* fe_b1   = (const float*)d_in[5];
  const float* fe_w2   = (const float*)d_in[6];
  const float* fe_b2   = (const float*)d_in[7];
  const float* ln1_g   = (const float*)d_in[8];
  const float* ln1_b   = (const float*)d_in[9];
  const float* wq      = (const float*)d_in[10];
  const float* wk      = (const float*)d_in[11];
  const float* wv      = (const float*)d_in[12];
  const float* w_rpe   = (const float*)d_in[13];
  const float* ow      = (const float*)d_in[14];
  const float* ob      = (const float*)d_in[15];
  const float* ln2_g   = (const float*)d_in[16];
  const float* ln2_b   = (const float*)d_in[17];
  const float* f1w     = (const float*)d_in[18];
  const float* f1b     = (const float*)d_in[19];
  const float* f2w     = (const float*)d_in[20];
  const float* f2b     = (const float*)d_in[21];
  const float* W_cat   = (const float*)d_in[22];
  const float* m1w     = (const float*)d_in[23];
  const float* m1b     = (const float*)d_in[24];
  const float* mlng    = (const float*)d_in[25];
  const float* mlnb    = (const float*)d_in[26];
  const float* m2w     = (const float*)d_in[27];
  const float* m2b     = (const float*)d_in[28];
  const float* pw      = (const float*)d_in[29];
  const float* pb      = (const float*)d_in[30];

  char* ws = (char*)d_ws;
  size_t off = 0;
  auto alloc = [&](size_t bytes) -> void* {
    void* p = ws + off;
    off += (bytes + 255) & ~(size_t)255;
    return p;
  };
  int*      ranks = (int*)     alloc(2 * NN * sizeof(int));
  int*      order = (int*)     alloc((size_t)NHASH * HH * NN * sizeof(int));
  float*    xpad  = (float*)   alloc((size_t)NN * 32 * sizeof(float));
  float*    xn    = (float*)   alloc((size_t)NN * DH * sizeof(float));
  float*    h0    = (float*)   alloc((size_t)NN * DH * sizeof(float));
  float*    h1    = (float*)   alloc((size_t)NN * DH * sizeof(float));
  float*    ffh   = (float*)   alloc((size_t)NN * DH * sizeof(float));
  float*    hcat  = (float*)   alloc((size_t)NN * CATW * sizeof(float));
  _Float16* q16   = (_Float16*)alloc((size_t)NN * HD * sizeof(_Float16));
  _Float16* k16   = (_Float16*)alloc((size_t)NN * HD * sizeof(_Float16));
  float*    vb    = (float*)   alloc((size_t)NN * HD * sizeof(float));
  float*    agg   = (float*)   alloc((size_t)NN * HD * sizeof(float));
  float*    omg   = (float*)   alloc(HH * MM * sizeof(float));
  float*    sqb   = (float*)   alloc((size_t)NN * HH * sizeof(float));
  float*    out_r = (float*)   alloc((size_t)NHASH * HH * NN * DH * sizeof(float));
  float*    lse_r = (float*)   alloc((size_t)NHASH * HH * NN * sizeof(float));
  float*    tbuf  = (float*)   alloc((size_t)NN * HALFW * sizeof(float));
  float*    t2    = (float*)   alloc((size_t)NN * HALFW * sizeof(float));
  float*    mhid  = (float*)   alloc((size_t)NN * MLPH * sizeof(float));
  float*    mhid2 = (float*)   alloc((size_t)NN * MLPH * sizeof(float));
  float*    accum = (float*)   alloc(HALFW * sizeof(float));
  _Float16* bpack = (_Float16*)alloc((size_t)64 * 1024 * sizeof(_Float16)); // max K*Nc = 64*512

  auto ggrid = [](int M, int Nc) { return ((M / 16) * (Nc / 16) + 3) / 4; };
  auto pgrid = [](int K, int Nc) { return ((Nc / 16) * (K / 32) * 32 + 255) / 256; };
  // pack + gemm pair (sequential on stream; deterministic, graph-capture safe)
  auto gemm = [&](const float* A, const float* B, const float* bias, const float* res,
                  void* C, int M, int K, int Nc, float alpha, int act, int outf16) {
    pack_b_kernel<<<pgrid(K, Nc), 256, 0, stream>>>(B, bpack, K, Nc);
    gemm_wmma<<<ggrid(M, Nc), 128, 0, stream>>>(A, bpack, bias, res, C,
                                                M, K, Nc, alpha, act, outf16);
  };

  const size_t SORT_LDS = (size_t)NN * sizeof(unsigned long long);  // 128 KB of 320 KB WGP LDS

  // --- bucket ordering (LDS-resident stable bitonic sorts) ---
  rank_sort_kernel <<<2, 1024, SORT_LDS, stream>>>(coords, ranks);
  order_sort_kernel<<<NHASH * HH, 1024, SORT_LDS, stream>>>(ranks, regions, order);

  // --- feature encoder (x and fe_w1 zero-padded to K=32 so the GEMM K-loop is exact) ---
  pad_x_kernel<<<(NN * 32 + 255) / 256, 256, 0, stream>>>(x, xpad);
  zero_kernel<<<(32 * DH + 255) / 256, 256, 0, stream>>>(mhid, 32 * DH);
  hipMemcpyAsync(mhid, fe_w1, (size_t)IN_DIM * DH * sizeof(float),
                 hipMemcpyDeviceToDevice, stream);
  gemm(xpad, mhid, fe_b1, nullptr, xn, NN, 32, DH, 1.f, ACT_RELU, 0);
  gemm(xn, fe_w2, fe_b2, nullptr, h0, NN, DH, DH, 1.f, ACT_NONE, 0);
  copy_cols<<<(NN * DH + 255) / 256, 256, 0, stream>>>(h0, hcat, 0);

  float* hc = h0;
  float* hn = h1;
  for (int l = 0; l < LL; ++l) {
    layernorm_kernel<<<NN / 8, 256, 0, stream>>>(hc, ln1_g + l * DH, ln1_b + l * DH,
                                                 xn, NN, DH, ACT_NONE);
    gemm(xn, wq + (size_t)l * DH * HD, nullptr, nullptr, q16, NN, DH, HD, 0.125f, ACT_NONE, 1);
    gemm(xn, wk + (size_t)l * DH * HD, nullptr, nullptr, k16, NN, DH, HD, 1.f, ACT_NONE, 1);
    gemm(xn, wv + (size_t)l * DH * HD, nullptr, nullptr, vb,  NN, DH, HD, 1.f, ACT_NONE, 0);
    omega_kernel<<<1, 128, 0, stream>>>(w_rpe + (size_t)l * HD * MM, omg);
    sq_kernel<<<(NN * HH + 255) / 256, 256, 0, stream>>>(coords, omg, sqb);
    attn_block_kernel<<<NHASH * HH * NB, 256, 0, stream>>>(q16, k16, vb, coords, omg,
                                                           sqb, order, out_r, lse_r);
    combine_rounds<<<(int)(((size_t)NN * HD + 255) / 256), 256, 0, stream>>>(out_r, lse_r, agg);
    gemm(agg, ow + (size_t)l * HD * DH, ob + l * DH, hc, hn, NN, HD, DH, 1.f, ACT_NONE, 0);
    { float* tp = hc; hc = hn; hn = tp; }
    layernorm_kernel<<<NN / 8, 256, 0, stream>>>(hc, ln2_g + l * DH, ln2_b + l * DH,
                                                 xn, NN, DH, ACT_NONE);
    gemm(xn, f1w + (size_t)l * DH * DH, f1b + l * DH, nullptr, ffh, NN, DH, DH, 1.f, ACT_RELU, 0);
    gemm(ffh, f2w + (size_t)l * DH * DH, f2b + l * DH, hc, hn, NN, DH, DH, 1.f, ACT_NONE, 0);
    { float* tp = hc; hc = hn; hn = tp; }
    copy_cols<<<(NN * DH + 255) / 256, 256, 0, stream>>>(hc, hcat, DH * (l + 1));
  }

  // --- output head ---
  gemm(hcat, W_cat, nullptr, nullptr, tbuf, NN, CATW, HALFW, 1.f, ACT_TANH, 0);
  gemm(tbuf, m1w, m1b, nullptr, mhid, NN, HALFW, MLPH, 1.f, ACT_NONE, 0);
  layernorm_kernel<<<NN / 8, 256, 0, stream>>>(mhid, mlng, mlnb, mhid2, NN, MLPH, ACT_TANH);
  gemm(mhid2, m2w, m2b, tbuf, t2, NN, MLPH, HALFW, 1.f, ACT_NONE, 0);
  zero_kernel<<<1, HALFW, 0, stream>>>(accum, HALFW);
  colsum_kernel<<<128, 256, 0, stream>>>(t2, accum);
  final_kernel<<<1, HALFW, 0, stream>>>(accum, pw, pb, (float*)d_out);
}
